// SHRED_62740882260324
// MI455X (gfx1250) — matrix operand order repaired and checked by reading, not compiled, not run on hardware
//
#include <hip/hip_runtime.h>
#include <hip/hip_bf16.h>

// ---------------------------------------------------------------------------
// Problem constants (from reference): B=1024, T=256, IN=H=64, 4H=256,
// DEC1=350 (pad 352), DEC2=400 (pad K to 416), OUT=4096.
// ---------------------------------------------------------------------------
#define RNN_T    256
#define RNN_B    1024
#define RROW     72          // padded row length (halves) for LDS staging
#define D1       350
#define D1P      352
#define D2       400
#define D2P      416
#define NOUT     4096

typedef __attribute__((ext_vector_type(16))) _Float16 v16h;
typedef __attribute__((ext_vector_type(8)))  _Float16 v8h;
typedef __attribute__((ext_vector_type(8)))  float    v8f;

// A-fragment (16x32 f16) from a row-major half buffer with row stride strideH.
// lanes 0-15: row = base+lane, halves = K{kk*32+0..7, kk*32+16..23}
// lanes 16-31: row = base+lane-16, halves = K{kk*32+8..15, kk*32+24..31}
__device__ __forceinline__ v16h frag_a(const _Float16* buf, int rowBase,
                                       int strideH, int lane, int kk) {
  const int lr  = lane & 15;
  const int off = (lane >> 4) * 8;
  const _Float16* p = buf + (size_t)(rowBase + lr) * strideH + kk * 32 + off;
  v8h lo = *(const v8h*)p;
  v8h hi = *(const v8h*)(p + 16);
  v16h a;
#pragma unroll
  for (int i = 0; i < 8; ++i) { a[i] = lo[i]; a[i + 8] = hi[i]; }
  return a;
}

// B-fragment (32x16) where B(k,n) = W[n][k], W row-major with stride strideH.
// lanes 0-15: col n = n0+lane, halves = W[n][kk*32+0..7, kk*32+16..23]
// lanes 16-31: col n = n0+lane-16, halves = W[n][kk*32+8..15, kk*32+24..31]
__device__ __forceinline__ v16h frag_b(const _Float16* w, int n, int strideH,
                                       int lane, int kk) {
  const int j = n * 16 + (lane & 15);
  const _Float16* p = w + (size_t)j * strideH + kk * 32 + ((lane >> 4) * 8);
  v8h lo = *(const v8h*)p;
  v8h hi = *(const v8h*)(p + 16);
  v16h b;
#pragma unroll
  for (int i = 0; i < 8; ++i) { b[i] = lo[i]; b[i + 8] = hi[i]; }
  return b;
}

__device__ __forceinline__ v8f wmma32(v16h a, v16h b, v8f c) {
  return __builtin_amdgcn_wmma_f32_16x16x32_f16(false, a, false, b,
                                                (short)0, c, false, false);
}
__device__ __forceinline__ float sigm(float v) { return 1.0f / (1.0f + __expf(-v)); }

// ---------------------------------------------------------------------------
// Persistent 2-layer LSTM. grid = B/64 blocks, 256 threads (8 waves).
// Wave w: M-tile m = w&3 (16 batch rows), gate-quadrant pairs q in
// {2*(w>>2), 2*(w>>2)+1}; owns gate tiles n = q + {0,4,8,12} so that
// i/f/g/o for each (row, j=16q+lane%16) land in the same lane/VGPR.
// ---------------------------------------------------------------------------
struct RnnLds {
  _Float16 wih0[256 * RROW];
  _Float16 whh0[256 * RROW];
  _Float16 wih1[256 * RROW];
  _Float16 whh1[256 * RROW];
  _Float16 xbuf[64 * RROW];
  _Float16 h0buf[64 * RROW];
  _Float16 h1buf[64 * RROW];
  float bias0[256];
  float bias1[256];
};

__global__ __launch_bounds__(256) void k_rnn(
    const float* __restrict__ x,
    const float* __restrict__ Wih0, const float* __restrict__ Whh0,
    const float* __restrict__ bih0, const float* __restrict__ bhh0,
    const float* __restrict__ Wih1, const float* __restrict__ Whh1,
    const float* __restrict__ bih1, const float* __restrict__ bhh1,
    _Float16* __restrict__ h1out)
{
  __shared__ RnnLds s;
  const int tid  = threadIdx.x;
  const int wave = tid >> 5;
  const int lane = tid & 31;
  const int b0   = blockIdx.x * 64;

  // Stage all four LSTM weight matrices (f32 -> f16) into LDS, padded rows.
  for (int i = tid; i < 256 * 64; i += 256) {
    int r = i >> 6, c = i & 63;
    s.wih0[r * RROW + c] = (_Float16)Wih0[i];
    s.whh0[r * RROW + c] = (_Float16)Whh0[i];
    s.wih1[r * RROW + c] = (_Float16)Wih1[i];
    s.whh1[r * RROW + c] = (_Float16)Whh1[i];
  }
  if (tid < 256) {
    s.bias0[tid] = bih0[tid] + bhh0[tid];
    s.bias1[tid] = bih1[tid] + bhh1[tid];
  }
  for (int i = tid; i < 64 * RROW; i += 256) {
    s.xbuf[i]  = (_Float16)0.0f;
    s.h0buf[i] = (_Float16)0.0f;
    s.h1buf[i] = (_Float16)0.0f;
  }
  __syncthreads();

  const int m    = wave & 3;
  const int qb   = (wave >> 2) * 2;
  const int lr   = lane & 15;
  const int lh   = lane >> 4;
  const int rowB = m * 16;

  float c0[2][8], c1[2][8];
#pragma unroll
  for (int q = 0; q < 2; ++q)
#pragma unroll
    for (int r = 0; r < 8; ++r) { c0[q][r] = 0.0f; c1[q][r] = 0.0f; }

  for (int t = 0; t < RNN_T; ++t) {
    // ---- stage x[:, t, :] tile (64 rows x 64 f32) -> f16 LDS -------------
    for (int i = tid; i < 64 * 16; i += 256) {
      int row = i >> 4, c4 = i & 15;
      float4 v = *(const float4*)(x + ((size_t)(b0 + row) * RNN_T + t) * 64 + c4 * 4);
      _Float16* d = &s.xbuf[row * RROW + c4 * 4];
      d[0] = (_Float16)v.x; d[1] = (_Float16)v.y;
      d[2] = (_Float16)v.z; d[3] = (_Float16)v.w;
    }
    __syncthreads();

    // ---- preload all A-fragments for this step (reads of x/h done here) --
    v16h ax0  = frag_a(s.xbuf,  rowB, RROW, lane, 0);
    v16h ax1  = frag_a(s.xbuf,  rowB, RROW, lane, 1);
    v16h ah00 = frag_a(s.h0buf, rowB, RROW, lane, 0);
    v16h ah01 = frag_a(s.h0buf, rowB, RROW, lane, 1);
    v16h ah10 = frag_a(s.h1buf, rowB, RROW, lane, 0);
    v16h ah11 = frag_a(s.h1buf, rowB, RROW, lane, 1);
    __syncthreads();

    // ---- layer 0: gates = x_t @ Wih0^T + h0 @ Whh0^T + b ------------------
#pragma unroll
    for (int qi = 0; qi < 2; ++qi) {
      const int q = qb + qi;
      v8f acc[4];
#pragma unroll
      for (int p = 0; p < 4; ++p) {
        const int n = q + 4 * p;
        v8f a = {};
        a = wmma32(ax0,  frag_b(s.wih0, n, RROW, lane, 0), a);
        a = wmma32(ax1,  frag_b(s.wih0, n, RROW, lane, 1), a);
        a = wmma32(ah00, frag_b(s.whh0, n, RROW, lane, 0), a);
        a = wmma32(ah01, frag_b(s.whh0, n, RROW, lane, 1), a);
        acc[p] = a;
      }
      const int j = q * 16 + lr;
#pragma unroll
      for (int r = 0; r < 8; ++r) {
        float iv = acc[0][r] + s.bias0[j];
        float fv = acc[1][r] + s.bias0[j + 64];
        float gv = acc[2][r] + s.bias0[j + 128];
        float ov = acc[3][r] + s.bias0[j + 192];
        float cc = sigm(fv) * c0[qi][r] + sigm(iv) * tanhf(gv);
        c0[qi][r] = cc;
        s.h0buf[(rowB + r + 8 * lh) * RROW + j] = (_Float16)(sigm(ov) * tanhf(cc));
      }
    }
    __syncthreads();
    v16h ab00 = frag_a(s.h0buf, rowB, RROW, lane, 0);
    v16h ab01 = frag_a(s.h0buf, rowB, RROW, lane, 1);

    // ---- layer 1: gates = h0_new @ Wih1^T + h1 @ Whh1^T + b ---------------
#pragma unroll
    for (int qi = 0; qi < 2; ++qi) {
      const int q = qb + qi;
      v8f acc[4];
#pragma unroll
      for (int p = 0; p < 4; ++p) {
        const int n = q + 4 * p;
        v8f a = {};
        a = wmma32(ab00, frag_b(s.wih1, n, RROW, lane, 0), a);
        a = wmma32(ab01, frag_b(s.wih1, n, RROW, lane, 1), a);
        a = wmma32(ah10, frag_b(s.whh1, n, RROW, lane, 0), a);
        a = wmma32(ah11, frag_b(s.whh1, n, RROW, lane, 1), a);
        acc[p] = a;
      }
      const int j = q * 16 + lr;
#pragma unroll
      for (int r = 0; r < 8; ++r) {
        float iv = acc[0][r] + s.bias1[j];
        float fv = acc[1][r] + s.bias1[j + 64];
        float gv = acc[2][r] + s.bias1[j + 128];
        float ov = acc[3][r] + s.bias1[j + 192];
        float cc = sigm(fv) * c1[qi][r] + sigm(iv) * tanhf(gv);
        c1[qi][r] = cc;
        s.h1buf[(rowB + r + 8 * lh) * RROW + j] = (_Float16)(sigm(ov) * tanhf(cc));
      }
    }
    __syncthreads();
  }

  for (int i = tid; i < 64 * 64; i += 256) {
    int r = i >> 6, c = i & 63;
    h1out[(size_t)(b0 + r) * 64 + c] = s.h1buf[r * RROW + c];
  }
}

// ---------------------------------------------------------------------------
// Weight prep: f32 -> zero-padded f16 copies in workspace.
//   w1p: 352x64  (rows 350..351 zero)
//   w2p: 400x352 (cols 350..351 zero)
//   w3p: 4096x416 (cols 400..415 zero)
// ---------------------------------------------------------------------------
__global__ void k_prep(const float* __restrict__ dW1, const float* __restrict__ dW2,
                       const float* __restrict__ dW3, _Float16* __restrict__ w1p,
                       _Float16* __restrict__ w2p, _Float16* __restrict__ w3p)
{
  size_t gid = (size_t)blockIdx.x * blockDim.x + threadIdx.x;
  size_t stride = (size_t)gridDim.x * blockDim.x;
  for (size_t i = gid; i < (size_t)D1P * 64; i += stride) {
    int n = (int)(i >> 6), k = (int)(i & 63);
    w1p[i] = (n < D1) ? (_Float16)dW1[n * 64 + k] : (_Float16)0.0f;
  }
  for (size_t i = gid; i < (size_t)D2 * D1P; i += stride) {
    int n = (int)(i / D1P), k = (int)(i % D1P);
    w2p[i] = (k < D1) ? (_Float16)dW2[n * D1 + k] : (_Float16)0.0f;
  }
  for (size_t i = gid; i < (size_t)NOUT * D2P; i += stride) {
    int n = (int)(i / D2P), k = (int)(i % D2P);
    w3p[i] = (k < D2) ? (_Float16)dW3[(size_t)n * D2 + k] : (_Float16)0.0f;
  }
}

// ---------------------------------------------------------------------------
// Decoder layer 1: y1 = relu(LN(h1 @ dW1^T + db1)).  grid = B/64 blocks.
// ---------------------------------------------------------------------------
struct Dec1Lds { _Float16 a[64 * RROW]; float z[64 * D1P]; };

__global__ __launch_bounds__(256) void k_dec1(
    const _Float16* __restrict__ h1, const _Float16* __restrict__ w1p,
    const float* __restrict__ db1, const float* __restrict__ g1,
    const float* __restrict__ be1, _Float16* __restrict__ y1)
{
  __shared__ Dec1Lds s;
  const int tid = threadIdx.x, wave = tid >> 5, lane = tid & 31;
  const int b0 = blockIdx.x * 64;
  for (int i = tid; i < 64 * 64; i += 256) {
    int r = i >> 6, c = i & 63;
    s.a[r * RROW + c] = h1[(size_t)(b0 + r) * 64 + c];
  }
  __syncthreads();
  for (int tt = wave; tt < 88; tt += 8) {          // 4 M-tiles x 22 N-tiles
    const int mm = tt & 3, n = tt >> 2;
    v8f acc = {};
    acc = wmma32(frag_a(s.a, mm * 16, RROW, lane, 0), frag_b(w1p, n, 64, lane, 0), acc);
    acc = wmma32(frag_a(s.a, mm * 16, RROW, lane, 1), frag_b(w1p, n, 64, lane, 1), acc);
    const int col = n * 16 + (lane & 15);
    const float bv = (col < D1) ? db1[col] : 0.0f;
#pragma unroll
    for (int r = 0; r < 8; ++r)
      s.z[(mm * 16 + r + 8 * (lane >> 4)) * D1P + col] = acc[r] + bv;
  }
  __syncthreads();
  if (tid < 64) {
    const float* zr = &s.z[tid * D1P];
    float mu = 0.0f;
    for (int k = 0; k < D1; ++k) mu += zr[k];
    mu *= (1.0f / D1);
    float var = 0.0f;
    for (int k = 0; k < D1; ++k) { float d = zr[k] - mu; var += d * d; }
    var *= (1.0f / D1);
    const float rs = rsqrtf(var + 1e-5f);
    _Float16* yr = y1 + (size_t)(b0 + tid) * D1P;
    for (int k = 0; k < D1P; ++k) {
      float v = 0.0f;
      if (k < D1) { v = (zr[k] - mu) * rs * g1[k] + be1[k]; v = v > 0.0f ? v : 0.0f; }
      yr[k] = (_Float16)v;
    }
  }
}

// ---------------------------------------------------------------------------
// Decoder layer 2: y2 = relu(LN(y1 @ dW2^T + db2)), K=352, N=400 (pad 416).
// ---------------------------------------------------------------------------
struct Dec2Lds { _Float16 a[64 * D1P]; float z[64 * D2]; };

__global__ __launch_bounds__(256) void k_dec2(
    const _Float16* __restrict__ y1, const _Float16* __restrict__ w2p,
    const float* __restrict__ db2, const float* __restrict__ g2,
    const float* __restrict__ be2, _Float16* __restrict__ y2)
{
  __shared__ Dec2Lds s;
  const int tid = threadIdx.x, wave = tid >> 5, lane = tid & 31;
  const int b0 = blockIdx.x * 64;
  for (int i = tid; i < 64 * (D1P / 8); i += 256) {
    int r = i / (D1P / 8), c = i % (D1P / 8);
    *(v8h*)&s.a[r * D1P + c * 8] = *(const v8h*)(y1 + (size_t)(b0 + r) * D1P + c * 8);
  }
  __syncthreads();
  for (int tt = wave; tt < 100; tt += 8) {         // 4 M-tiles x 25 N-tiles
    const int mm = tt & 3, n = tt >> 2;
    v8f acc = {};
    for (int kk = 0; kk < D1P / 32; ++kk)
      acc = wmma32(frag_a(s.a, mm * 16, D1P, lane, kk),
                   frag_b(w2p, n, D1P, lane, kk), acc);
    const int col = n * 16 + (lane & 15);
    const float bv = db2[col];
#pragma unroll
    for (int r = 0; r < 8; ++r)
      s.z[(mm * 16 + r + 8 * (lane >> 4)) * D2 + col] = acc[r] + bv;
  }
  __syncthreads();
  if (tid < 64) {
    const float* zr = &s.z[tid * D2];
    float mu = 0.0f;
    for (int k = 0; k < D2; ++k) mu += zr[k];
    mu *= (1.0f / D2);
    float var = 0.0f;
    for (int k = 0; k < D2; ++k) { float d = zr[k] - mu; var += d * d; }
    var *= (1.0f / D2);
    const float rs = rsqrtf(var + 1e-5f);
    _Float16* yr = y2 + (size_t)(b0 + tid) * D2P;
    for (int k = 0; k < D2P; ++k) {
      float v = 0.0f;
      if (k < D2) { v = (zr[k] - mu) * rs * g2[k] + be2[k]; v = v > 0.0f ? v : 0.0f; }
      yr[k] = (_Float16)v;
    }
  }
}

// ---------------------------------------------------------------------------
// Decoder layer 3: out = y2 @ dW3^T + db3. grid = (B/64, 4096/256).
// ---------------------------------------------------------------------------
struct Dec3Lds { _Float16 a[64 * D2P]; };

__global__ __launch_bounds__(256) void k_dec3(
    const _Float16* __restrict__ y2, const _Float16* __restrict__ w3p,
    const float* __restrict__ db3, float* __restrict__ out)
{
  __shared__ Dec3Lds s;
  const int tid = threadIdx.x, wave = tid >> 5, lane = tid & 31;
  const int b0 = blockIdx.x * 64;
  const int n0 = blockIdx.y * 256;
  for (int i = tid; i < 64 * (D2P / 8); i += 256) {
    int r = i / (D2P / 8), c = i % (D2P / 8);
    *(v8h*)&s.a[r * D2P + c * 8] = *(const v8h*)(y2 + (size_t)(b0 + r) * D2P + c * 8);
  }
  __syncthreads();
  for (int tt = wave; tt < 64; tt += 8) {          // 4 M-tiles x 16 N-tiles
    const int mm = tt & 3, nl = tt >> 2;
    v8f acc = {};
    for (int kk = 0; kk < D2P / 32; ++kk)
      acc = wmma32(frag_a(s.a, mm * 16, D2P, lane, kk),
                   frag_b(w3p + (size_t)n0 * D2P, nl, D2P, lane, kk), acc);
    const int col = n0 + nl * 16 + (lane & 15);
    const float bv = db3[col];
#pragma unroll
    for (int r = 0; r < 8; ++r)
      out[(size_t)(b0 + mm * 16 + r + 8 * (lane >> 4)) * NOUT + col] = acc[r] + bv;
  }
}

// ---------------------------------------------------------------------------
extern "C" void kernel_launch(void* const* d_in, const int* in_sizes, int n_in,
                              void* d_out, int out_size, void* d_ws, size_t ws_size,
                              hipStream_t stream) {
  (void)in_sizes; (void)n_in; (void)out_size; (void)ws_size;
  const float* x     = (const float*)d_in[0];
  const float* Wih0  = (const float*)d_in[1];
  const float* Whh0  = (const float*)d_in[2];
  const float* bih0  = (const float*)d_in[3];
  const float* bhh0  = (const float*)d_in[4];
  const float* Wih1  = (const float*)d_in[5];
  const float* Whh1  = (const float*)d_in[6];
  const float* bih1  = (const float*)d_in[7];
  const float* bhh1  = (const float*)d_in[8];
  const float* dW1   = (const float*)d_in[9];
  const float* db1   = (const float*)d_in[10];
  const float* g1    = (const float*)d_in[11];
  const float* be1   = (const float*)d_in[12];
  const float* dW2   = (const float*)d_in[13];
  const float* db2   = (const float*)d_in[14];
  const float* g2    = (const float*)d_in[15];
  const float* be2   = (const float*)d_in[16];
  const float* dW3   = (const float*)d_in[17];
  const float* db3   = (const float*)d_in[18];
  float* out = (float*)d_out;

  char* ws = (char*)d_ws;
  size_t off = 0;
  auto take = [&](size_t bytes) { void* p = ws + off; off += (bytes + 255) & ~(size_t)255; return p; };
  _Float16* h1f = (_Float16*)take((size_t)RNN_B * 64 * 2);
  _Float16* y1  = (_Float16*)take((size_t)RNN_B * D1P * 2);
  _Float16* y2  = (_Float16*)take((size_t)RNN_B * D2P * 2);
  _Float16* w1p = (_Float16*)take((size_t)D1P * 64 * 2);
  _Float16* w2p = (_Float16*)take((size_t)D2 * D1P * 2);
  _Float16* w3p = (_Float16*)take((size_t)NOUT * D2P * 2);

  k_prep<<<1024, 256, 0, stream>>>(dW1, dW2, dW3, w1p, w2p, w3p);
  k_rnn<<<RNN_B / 64, 256, 0, stream>>>(x, Wih0, Whh0, bih0, bhh0,
                                        Wih1, Whh1, bih1, bhh1, h1f);
  k_dec1<<<RNN_B / 64, 256, 0, stream>>>(h1f, w1p, db1, g1, be1, y1);
  k_dec2<<<RNN_B / 64, 256, 0, stream>>>(y1, w2p, db2, g2, be2, y2);
  k_dec3<<<dim3(RNN_B / 64, NOUT / 256), 256, 0, stream>>>(y2, w3p, db3, out);
}
